// BayesModel_75737453297749
// MI455X (gfx1250) — matrix operand order repaired and checked by reading, not compiled, not run on hardware
//
#include <hip/hip_runtime.h>

typedef unsigned int   u32;
typedef unsigned short u16;

typedef __attribute__((ext_vector_type(16))) __bf16 v16bf;
typedef __attribute__((ext_vector_type(8)))  float  v8f;
typedef __attribute__((ext_vector_type(4)))  u32    u32x4;
typedef __attribute__((ext_vector_type(4)))  float  f32x4;

// gfx1250 async-LDS builtin takes v4i pointers in AS1 (global) / AS3 (LDS)
typedef int i32x4 __attribute__((vector_size(16)));
typedef __attribute__((address_space(1))) i32x4 as1_i32x4;
typedef __attribute__((address_space(3))) i32x4 as3_i32x4;

#define CC     256      // output channels (M)
#define CIN    1024     // input channels  (K)
#define NB     16       // batch
#define HW     9216     // pixels per image (N total)
#define BN     64       // pixels per block
#define KC     32       // K chunk (one bf16 WMMA depth)
#define AVROW  5        // u32x4 vectors per A LDS row (64B data + 16B pad)
#define BVROW  5        // u32x4 vectors per B LDS row
#define XPITCH 65       // floats per xc LDS row (64 + 1 pad)
#define ALPHA_C 0.8f
#define BETA_C  0.8f

#if __has_builtin(__builtin_amdgcn_global_load_async_to_lds_b128)
#define HAVE_ASYNC_LDS 1
#endif

__device__ __forceinline__ u16 f2bf(float f) {
  // round-to-nearest-even fp32 -> bf16
  u32 u = __float_as_uint(f);
  u32 r = u + 0x7FFFu + ((u >> 16) & 1u);
  return (u16)(r >> 16);
}

union Frag {
  struct { u32x4 lo, hi; } u;
  v16bf v;
};

__device__ __forceinline__ v8f wmma_bf16(v16bf a, v16bf b, v8f c) {
  // (neg_a, A, neg_b, B, c_mod, C, reuse_a, reuse_b)
  return __builtin_amdgcn_wmma_f32_16x16x32_bf16(false, a, false, b, (short)0, c,
                                                 false, false);
}

// copy one 64-byte bf16 W row segment global->LDS
__device__ __forceinline__ void copy_row64(const u16* gsrc, u32x4* ldst) {
#ifdef HAVE_ASYNC_LDS
  // INST_OFFSET is added to both the global and LDS addresses (ISA §15.18)
  __builtin_amdgcn_global_load_async_to_lds_b128((as1_i32x4*)gsrc, (as3_i32x4*)ldst, 0, 0);
  __builtin_amdgcn_global_load_async_to_lds_b128((as1_i32x4*)gsrc, (as3_i32x4*)ldst, 16, 0);
  __builtin_amdgcn_global_load_async_to_lds_b128((as1_i32x4*)gsrc, (as3_i32x4*)ldst, 32, 0);
  __builtin_amdgcn_global_load_async_to_lds_b128((as1_i32x4*)gsrc, (as3_i32x4*)ldst, 48, 0);
#else
  const u32x4* s = (const u32x4*)gsrc;
  ldst[0] = s[0]; ldst[1] = s[1]; ldst[2] = s[2]; ldst[3] = s[3];
#endif
}

__device__ __forceinline__ void wait_async_lds() {
#ifdef HAVE_ASYNC_LDS
#if __has_builtin(__builtin_amdgcn_s_wait_asynccnt)
  __builtin_amdgcn_s_wait_asynccnt(0);
#else
  asm volatile("s_wait_asynccnt 0" ::: "memory");
#endif
#endif
}

// ---------------------------------------------------------------- prep ------
__global__ __launch_bounds__(256) void prep_kernel(const float* __restrict__ W,
                                                   u16* __restrict__ wbf,
                                                   float* __restrict__ accum) {
  int idx = blockIdx.x * 256 + threadIdx.x;
  if (idx < CC * CIN) wbf[idx] = f2bf(W[idx]);
  if (idx < 32 + 2 * NB * CC) accum[idx] = 0.0f;  // nt_fg[16] nt_bg[16] sum_fg sum_bg
}

// ---------------------------------------------------------------- main ------
__global__ __launch_bounds__(256) void bayes_main(
    const float* __restrict__ x, const u16* __restrict__ wbf,
    const float* __restrict__ n_fg, const float* __restrict__ n_bg,
    const float* __restrict__ s_fg, const float* __restrict__ s_bg,
    float* __restrict__ out_post,
    float* __restrict__ nt_fg, float* __restrict__ nt_bg,
    float* __restrict__ sum_fg, float* __restrict__ sum_bg) {
  __shared__ union {
    struct {                                   // GEMM phase, double buffered (50 KB)
      u32x4 As[2][CC * AVROW];
      u32x4 Bs[2][BN * BVROW];
    } g;
    struct { float xc[CC * XPITCH]; } e;       // epilogue (65 KB)
  } sm;
  __shared__ float sfg[CC], sbg[CC];
  __shared__ float mmred[2][4][BN];
  __shared__ float postbuf[2][BN];

  const int tid   = threadIdx.x;
  const int b     = blockIdx.y;
  const int n0    = blockIdx.x * BN;
  const int lane  = tid & 31;
  const int wave  = tid >> 5;
  const int mwave = (wave >> 1) * 64;   // 4 waves along M (256 rows)
  const int nwave = (wave & 1) * 32;    // 2 waves along N (64 pixels)

  sfg[tid] = s_fg[b * CC + tid];
  sbg[tid] = s_bg[b * CC + tid];

  v8f acc[4][2] = {};

  // B-tile loader mapping: thread -> (k-pair, 4-pixel group)
  const int kp = tid >> 4;   // 0..15
  const int pg = tid & 15;   // 0..15
  const float* xb = x + (size_t)b * CIN * HW + n0 + pg * 4;
  const u16*   wrow = wbf + (size_t)tid * CIN;   // A loader: thread == channel row

  // ---- prologue: fill buffer 0 with chunk kt=0
  {
    copy_row64(wrow, &sm.g.As[0][tid * AVROW]);                  // async A copy
    const float* xsrc = xb + (size_t)(2 * kp) * HW;
    f32x4 r0 = *(const f32x4*)xsrc;
    f32x4 r1 = *(const f32x4*)(xsrc + HW);
    u32* bs = (u32*)sm.g.Bs[0];
    int nrow = pg * 4;
    bs[(nrow + 0) * (BVROW * 4) + kp] = (u32)f2bf(r0.x) | ((u32)f2bf(r1.x) << 16);
    bs[(nrow + 1) * (BVROW * 4) + kp] = (u32)f2bf(r0.y) | ((u32)f2bf(r1.y) << 16);
    bs[(nrow + 2) * (BVROW * 4) + kp] = (u32)f2bf(r0.z) | ((u32)f2bf(r1.z) << 16);
    bs[(nrow + 3) * (BVROW * 4) + kp] = (u32)f2bf(r0.w) | ((u32)f2bf(r1.w) << 16);
    wait_async_lds();
  }
  __syncthreads();

  for (int kt = 0; kt < CIN / KC; ++kt) {
    const int  buf      = kt & 1;
    const bool has_next = (kt + 1) < (CIN / KC);
    f32x4 r0, r1;
    if (has_next) {
      const int k0n = (kt + 1) * KC;
      copy_row64(wrow + k0n, &sm.g.As[buf ^ 1][tid * AVROW]);    // async A, next buf
      const float* xsrc = xb + (size_t)(k0n + 2 * kp) * HW;
      r0 = *(const f32x4*)xsrc;
      r1 = *(const f32x4*)(xsrc + HW);
    }

    // ---- fragments + 8 WMMAs on current buffer
    const u32x4* Asb = sm.g.As[buf];
    const u32x4* Bsb = sm.g.Bs[buf];
    Frag fb0, fb1;
    {
      int brow = nwave + (lane & 15);
      int bv   = (lane >> 4) * 2;               // lane-half: K 0-15 / 16-31
      fb0.u.lo = Bsb[brow * BVROW + bv];
      fb0.u.hi = Bsb[brow * BVROW + bv + 1];
      fb1.u.lo = Bsb[(brow + 16) * BVROW + bv];
      fb1.u.hi = Bsb[(brow + 16) * BVROW + bv + 1];
    }
#pragma unroll
    for (int tm = 0; tm < 4; ++tm) {
      int arow = mwave + tm * 16 + (lane & 15);
      int av   = (lane >> 4) * 2;               // lane-half: K 0-15 / 16-31
      Frag fa;
      fa.u.lo = Asb[arow * AVROW + av];
      fa.u.hi = Asb[arow * AVROW + av + 1];
      acc[tm][0] = wmma_bf16(fa.v, fb0.v, acc[tm][0]);
      acc[tm][1] = wmma_bf16(fa.v, fb1.v, acc[tm][1]);
    }

    if (has_next) {
      // ---- convert + store B into next buffer, then fence async A copy
      u32* bs = (u32*)sm.g.Bs[buf ^ 1];
      int nrow = pg * 4;
      bs[(nrow + 0) * (BVROW * 4) + kp] = (u32)f2bf(r0.x) | ((u32)f2bf(r1.x) << 16);
      bs[(nrow + 1) * (BVROW * 4) + kp] = (u32)f2bf(r0.y) | ((u32)f2bf(r1.y) << 16);
      bs[(nrow + 2) * (BVROW * 4) + kp] = (u32)f2bf(r0.z) | ((u32)f2bf(r1.z) << 16);
      bs[(nrow + 3) * (BVROW * 4) + kp] = (u32)f2bf(r0.w) | ((u32)f2bf(r1.w) << 16);
      wait_async_lds();
    }
    __syncthreads();
  }

  // ---- stage 1: spill accumulators (full 256ch x 64px xc tile) to LDS
#pragma unroll
  for (int tm = 0; tm < 4; ++tm)
#pragma unroll
    for (int tn = 0; tn < 2; ++tn)
#pragma unroll
      for (int v = 0; v < 8; ++v) {
        int m = mwave + tm * 16 + (lane >> 4) * 8 + v;  // C/D layout
        int n = nwave + tn * 16 + (lane & 15);
        sm.e.xc[m * XPITCH + n] = acc[tm][tn][v];
      }
  __syncthreads();

  // ---- stage 2a: per-pixel partial squared distances (4 threads / pixel)
  {
    int q = tid >> 6, n = tid & 63;
    float mf = 0.0f, mb = 0.0f;
    for (int i = 0; i < 64; ++i) {
      int m = q * 64 + i;
      float v = sm.e.xc[m * XPITCH + n];
      float df = v - sfg[m]; mf += df * df;
      float db = v - sbg[m]; mb += db * db;
    }
    mmred[0][q][n] = mf;
    mmred[1][q][n] = mb;
  }
  __syncthreads();

  // ---- stage 2b: posteriors (literal reference math, incl. overflow behavior)
  if (tid < BN) {
    int n = tid;
    float mf = mmred[0][0][n] + mmred[0][1][n] + mmred[0][2][n] + mmred[0][3][n];
    float mb = mmred[1][0][n] + mmred[1][1][n] + mmred[1][2][n] + mmred[1][3][n];
    float nf = n_fg[b], nb = n_bg[b];
    float pf = nf / (nf + nb);
    float llf = 2.0f / (1.0f + expf(mf));
    float llb = 2.0f / (1.0f + expf(mb));
    float numf = pf * llf;
    float numb = (1.0f - pf) * llb;
    float den  = numf + numb;
    float postf = numf / den;
    float postb = numb / den;
    out_post[(size_t)b * HW + n0 + n] = postf;
    postbuf[0][n] = postf;
    postbuf[1][n] = postb;
  }
  __syncthreads();

  // ---- stage 3: per-channel  sum_n xc * post  (one thread per channel)
  {
    int m = tid;
    float sf = 0.0f, sb = 0.0f;
    for (int n = 0; n < 64; ++n) {
      float v = sm.e.xc[m * XPITCH + n];
      sf += v * postbuf[0][n];
      sb += v * postbuf[1][n];
    }
    atomicAdd(&sum_fg[b * CC + m], sf);
    atomicAdd(&sum_bg[b * CC + m], sb);
  }
  if (tid == 0) {
    float s = 0.0f;
    for (int n = 0; n < 64; ++n) s += postbuf[0][n];
    atomicAdd(&nt_fg[b], s);
  }
  if (tid == 1) {
    float s = 0.0f;
    for (int n = 0; n < 64; ++n) s += postbuf[1][n];
    atomicAdd(&nt_bg[b], s);
  }
}

// ------------------------------------------------------------- finalize -----
__global__ __launch_bounds__(256) void finalize_kernel(
    const float* __restrict__ n_fg, const float* __restrict__ n_bg,
    const float* __restrict__ s_fg, const float* __restrict__ s_bg,
    const float* __restrict__ nt_fg, const float* __restrict__ nt_bg,
    const float* __restrict__ sum_fg, const float* __restrict__ sum_bg,
    float* __restrict__ out) {
  int b = blockIdx.x, m = threadIdx.x;
  float ntf = nt_fg[b], ntb = nt_bg[b];
  float* out_nf = out + (size_t)NB * HW;   // 147456
  float* out_nb = out_nf + NB;             // 147472
  float* out_sf = out_nb + NB;             // 147488
  float* out_sb = out_sf + NB * CC;        // 151584
  out_sf[b * CC + m] = BETA_C * s_fg[b * CC + m] + sum_fg[b * CC + m] / ntf;
  out_sb[b * CC + m] = BETA_C * s_bg[b * CC + m] + sum_bg[b * CC + m] / ntb;
  if (m == 0) out_nf[b] = ALPHA_C * n_fg[b] + ntf;
  if (m == 1) out_nb[b] = ALPHA_C * n_bg[b] + ntb;
}

// --------------------------------------------------------------- launch -----
extern "C" void kernel_launch(void* const* d_in, const int* in_sizes, int n_in,
                              void* d_out, int out_size, void* d_ws, size_t ws_size,
                              hipStream_t stream) {
  (void)in_sizes; (void)n_in; (void)out_size; (void)ws_size;
  const float* x    = (const float*)d_in[0];
  const float* W    = (const float*)d_in[1];
  const float* n_fg = (const float*)d_in[2];
  const float* n_bg = (const float*)d_in[3];
  const float* s_fg = (const float*)d_in[4];
  const float* s_bg = (const float*)d_in[5];
  float* out = (float*)d_out;

  // workspace: [ W as bf16 : 512KB ][ nt_fg 16 | nt_bg 16 | sum_fg 4096 | sum_bg 4096 ]
  u16*   wbf    = (u16*)d_ws;
  float* accum  = (float*)((char*)d_ws + (size_t)CC * CIN * sizeof(u16));
  float* nt_fg  = accum;
  float* nt_bg  = accum + NB;
  float* sum_fg = accum + 2 * NB;
  float* sum_bg = accum + 2 * NB + NB * CC;

  prep_kernel<<<dim3((CC * CIN) / 256), 256, 0, stream>>>(W, wbf, accum);
  bayes_main<<<dim3(HW / BN, NB), 256, 0, stream>>>(
      x, wbf, n_fg, n_bg, s_fg, s_bg, out, nt_fg, nt_bg, sum_fg, sum_bg);
  finalize_kernel<<<dim3(NB), 256, 0, stream>>>(
      n_fg, n_bg, s_fg, s_bg, nt_fg, nt_bg, sum_fg, sum_bg, out);
}
